// GraphConvLayer_8031588844218
// MI455X (gfx1250) — matrix operand order repaired and checked by reading, not compiled, not run on hardware
//
#include <hip/hip_runtime.h>
#include <cstdint>

// ---------------------------------------------------------------------------
// CDNA5 (gfx1250) GraphConv layer: BN folded into bf16 weights, all dense
// math on v_wmma_f32_16x16x32_bf16, segment-mean via global f32 atomics.
// Gelu: hardware v_tanh_f32 (gfx1250) tanh-form if available, else
// branch-free A&S erf approximation.
// ---------------------------------------------------------------------------

typedef __attribute__((ext_vector_type(16))) __bf16 v16bf;
typedef __attribute__((ext_vector_type(8)))  float  v8f;

#define BN_EPS 1e-3f

#if defined(__has_builtin)
#if __has_builtin(__builtin_amdgcn_tanhf)
#define GELU_USE_HW_TANH 1
#endif
#endif

__device__ __forceinline__ float gelu_fast(float v) {
#ifdef GELU_USE_HW_TANH
  // tanh-form gelu on hardware v_tanh_f32: ~5 VALU + 1 TRANS.
  const float u = 0.79788456080286535588f *
                  __builtin_fmaf(0.044715f * v * v, v, v);
  const float t = __builtin_amdgcn_tanhf(u);
  const float h = 0.5f * v;
  return __builtin_fmaf(h, t, h);
#else
  // Branch-free erf (A&S 7.1.26), |err| < 1.5e-7.
  const float z  = v * 0.70710678118654752440f;
  const float az = __builtin_fabsf(z);
  const float t  = __builtin_amdgcn_rcpf(__builtin_fmaf(0.3275911f, az, 1.0f));
  float p = __builtin_fmaf(1.061405429f, t, -1.453152027f);
  p = __builtin_fmaf(p, t, 1.421413741f);
  p = __builtin_fmaf(p, t, -0.284496736f);
  p = __builtin_fmaf(p, t, 0.254829592f);
  p *= t;
  const float e = __expf(-az * az);
  float erfv = __builtin_fmaf(-p, e, 1.0f);
  erfv = __builtin_copysignf(erfv, z);
  return 0.5f * v * (1.0f + erfv);
#endif
}

__device__ __forceinline__ v8f wmma_bf16(v16bf a, v16bf b, v8f c) {
  return __builtin_amdgcn_wmma_f32_16x16x32_bf16(false, a, false, b,
                                                 (short)0, c, false, false);
}

// B fragment (32x16 bf16, K x N): lane = col (lane&15);
// element j -> K = (j&7) + 16*(j>>3) + 8*(lane>>4).
__device__ __forceinline__ void load_bfrag(v16bf& fr, const __bf16* W,
                                           int k0, int nt, int lane) {
  const int n  = (lane & 15) + nt * 16;
  const int kb = k0 + ((lane >> 4) << 3);
#pragma unroll
  for (int j = 0; j < 16; ++j) fr[j] = W[(kb + (j & 7) + ((j >> 3) << 4)) * 64 + n];
}

// A fragment (16x32 bf16) from an f32 row: lane = row (lane&15);
// element j -> K = (j&7) + 16*(j>>3) + 8*(lane>>4)  (ISA 16-bit A layout).
__device__ __forceinline__ void load_afrag_f32(v16bf& fr, const float* row,
                                               int hi, int koff, float scale) {
  const float* p0 = row + koff + 8 * hi;
  const float* p1 = row + koff + 16 + 8 * hi;
#pragma unroll
  for (int j = 0; j < 8; ++j) fr[j]     = (__bf16)(p0[j] * scale);
#pragma unroll
  for (int j = 0; j < 8; ++j) fr[8 + j] = (__bf16)(p1[j] * scale);
}

__device__ __forceinline__ void load_afrag_lds(v16bf& fr, const __bf16* row,
                                               int hi, int koff) {
#pragma unroll
  for (int j = 0; j < 8; ++j) fr[j]     = row[koff + 8 * hi + j];
#pragma unroll
  for (int j = 0; j < 8; ++j) fr[8 + j] = row[koff + 16 + 8 * hi + j];
}

// ---------------------------------------------------------------------------
// Prep: fold BN (inference affine) into Dense weights -> bf16 + f32 bias.
//   BN(x) @ W + c = x @ (diag(a) W) + (c + b' W),  a=g*rsqrt(v+eps), b'=b-a*m
// ---------------------------------------------------------------------------
__global__ void prep_kernel(const float* pbn1, const float* pw1, const float* pc1,
                            const float* pbn2, const float* pw2, const float* pc2,
                            const float* ubn1, const float* uw1, const float* uc1,
                            const float* ubn2, const float* uw2, const float* uc2,
                            __bf16* W1f, float* c1f, __bf16* W2f, float* c2f,
                            __bf16* U1f, float* u1cf, __bf16* U2f, float* u2cf) {
  const int t = threadIdx.x;
  const float *bn, *W, *c; int fan, n; __bf16* Wf; float* cf;
  if (t < 64)        { bn = pbn1; W = pw1; c = pc1; fan = 64;  Wf = W1f; cf = c1f;  n = t; }
  else if (t < 128)  { bn = pbn2; W = pw2; c = pc2; fan = 64;  Wf = W2f; cf = c2f;  n = t - 64; }
  else if (t < 192)  { bn = ubn1; W = uw1; c = uc1; fan = 128; Wf = U1f; cf = u1cf; n = t - 128; }
  else               { bn = ubn2; W = uw2; c = uc2; fan = 64;  Wf = U2f; cf = u2cf; n = t - 192; }
  float acc = c[n];
  for (int k = 0; k < fan; ++k) {
    const float a   = bn[k] * rsqrtf(bn[3 * fan + k] + BN_EPS);
    const float wkn = W[k * 64 + n];
    Wf[k * 64 + n] = (__bf16)(a * wkn);
    acc += (bn[fan + k] - a * bn[2 * fan + k]) * wkn;
  }
  cf[n] = acc;
}

__global__ void init_kernel(float* s, float* cnt, long long n64, long long n) {
  const long long stride = (long long)gridDim.x * blockDim.x;
  for (long long i = (long long)blockIdx.x * blockDim.x + threadIdx.x; i < n64; i += stride)
    s[i] = 0.0f;
  for (long long i = (long long)blockIdx.x * blockDim.x + threadIdx.x; i < n; i += stride)
    cnt[i] = 0.0f;
}

// ---------------------------------------------------------------------------
// Edge kernel: one wave per 16-edge tile (grid-stride).
//   gather x[nbr] -> GEMM1(gelu) -> LDS relayout -> GEMM2(gelu) -> *ew
//   -> atomicAdd into s[node], cnt[node].
// ---------------------------------------------------------------------------
__global__ __launch_bounds__(256)
void edge_kernel(const float* __restrict__ x, const int* __restrict__ edges,
                 const float* __restrict__ ew,
                 const __bf16* __restrict__ W1f, const float* __restrict__ c1f,
                 const __bf16* __restrict__ W2f, const float* __restrict__ c2f,
                 float* __restrict__ s, float* __restrict__ cnt, int E) {
  __shared__ __bf16 hbuf[8][1024];   // per-wave 16x64 bf16 staging tile
  __shared__ int    ndbuf[8][16];
  __shared__ float  ewbuf[8][16];

  const int lane = threadIdx.x & 31;
  const int wv   = threadIdx.x >> 5;
  const int m    = lane & 15;
  const int hi   = lane >> 4;
  const int numTiles = (E + 15) >> 4;
  const int waveId   = blockIdx.x * 8 + wv;
  const int nWaves   = gridDim.x * 8;

  v16bf b1[8], b2[8];                // [kc*4+nt], kc=K/32, nt=N/16
  float bias1[4], bias2[4];
#pragma unroll
  for (int kc = 0; kc < 2; ++kc)
#pragma unroll
    for (int nt = 0; nt < 4; ++nt) {
      load_bfrag(b1[kc * 4 + nt], W1f, kc * 32, nt, lane);
      load_bfrag(b2[kc * 4 + nt], W2f, kc * 32, nt, lane);
    }
#pragma unroll
  for (int nt = 0; nt < 4; ++nt) {
    bias1[nt] = c1f[nt * 16 + m];
    bias2[nt] = c2f[nt * 16 + m];
  }

  for (int tile = waveId; tile < numTiles; tile += nWaves) {
    const int  eidx  = tile * 16 + m;
    const bool valid = eidx < E;
    const int  nbr   = valid ? edges[E + eidx] : 0;
    if (hi == 0) {
      ndbuf[wv][m] = valid ? edges[eidx] : -1;
      ewbuf[wv][m] = valid ? ew[eidx] : 0.0f;
    }
    const float* xr = x + (long long)nbr * 64;
    v16bf a0, a1;
    load_afrag_f32(a0, xr, hi, 0, 1.0f);
    load_afrag_f32(a1, xr, hi, 32, 1.0f);

    // layer 1 (BN pre-folded) + gelu -> LDS
#pragma unroll
    for (int nt = 0; nt < 4; ++nt) {
      v8f c = {};
      c = wmma_bf16(a1, b1[4 + nt], c);
      c = wmma_bf16(a0, b1[nt], c);
#pragma unroll
      for (int r = 0; r < 8; ++r)
        hbuf[wv][(r + 8 * hi) * 64 + nt * 16 + m] =
            (__bf16)gelu_fast(c[r] + bias1[nt]);
    }
    asm volatile("s_wait_dscnt 0" ::: "memory");  // wave-private LDS RAW

    v16bf a20, a21;
    load_afrag_lds(a20, &hbuf[wv][m * 64], hi, 0);
    load_afrag_lds(a21, &hbuf[wv][m * 64], hi, 32);

    // layer 2 + gelu, scale by ew, scatter-add per node
#pragma unroll
    for (int nt = 0; nt < 4; ++nt) {
      v8f c = {};
      c = wmma_bf16(a21, b2[4 + nt], c);
      c = wmma_bf16(a20, b2[nt], c);
#pragma unroll
      for (int r = 0; r < 8; ++r) {
        const int row  = r + 8 * hi;
        const int node = ndbuf[wv][row];
        if (node >= 0) {
          const float msg = gelu_fast(c[r] + bias2[nt]) * ewbuf[wv][row];
          atomicAdd(&s[(long long)node * 64 + nt * 16 + m], msg);
        }
      }
    }
    if (hi == 0) {
      const int node = ndbuf[wv][m];
      if (node >= 0) atomicAdd(&cnt[node], 1.0f);
    }
  }
}

// ---------------------------------------------------------------------------
// Node kernel: one wave per 16-node tile.
//   A = [x | s/max(cnt,1)] (16x128) -> GEMM(128->64, gelu) -> LDS relayout
//   -> GEMM(64->64, gelu) -> out.
// ---------------------------------------------------------------------------
__global__ __launch_bounds__(256)
void node_kernel(const float* __restrict__ x, const float* __restrict__ s,
                 const float* __restrict__ cnt,
                 const __bf16* __restrict__ U1f, const float* __restrict__ u1cf,
                 const __bf16* __restrict__ U2f, const float* __restrict__ u2cf,
                 float* __restrict__ out, int N) {
  __shared__ __bf16 hbuf[8][1024];

  const int lane = threadIdx.x & 31;
  const int wv   = threadIdx.x >> 5;
  const int m    = lane & 15;
  const int hi   = lane >> 4;
  const int numTiles = (N + 15) >> 4;
  const int waveId   = blockIdx.x * 8 + wv;
  const int nWaves   = gridDim.x * 8;

  v16bf bu1[16], bu2[8];
  float bias1[4], bias2[4];
#pragma unroll
  for (int kc = 0; kc < 4; ++kc)
#pragma unroll
    for (int nt = 0; nt < 4; ++nt)
      load_bfrag(bu1[kc * 4 + nt], U1f, kc * 32, nt, lane);
#pragma unroll
  for (int kc = 0; kc < 2; ++kc)
#pragma unroll
    for (int nt = 0; nt < 4; ++nt)
      load_bfrag(bu2[kc * 4 + nt], U2f, kc * 32, nt, lane);
#pragma unroll
  for (int nt = 0; nt < 4; ++nt) {
    bias1[nt] = u1cf[nt * 16 + m];
    bias2[nt] = u2cf[nt * 16 + m];
  }

  for (int tile = waveId; tile < numTiles; tile += nWaves) {
    const int   row = tile * 16 + m;
    const int   rr  = (row < N) ? row : 0;
    const float inv = 1.0f / fmaxf(cnt[rr], 1.0f);
    const float* xr = x + (long long)rr * 64;
    const float* sr = s + (long long)rr * 64;

    v16bf a[4];
    load_afrag_f32(a[0], xr, hi, 0, 1.0f);
    load_afrag_f32(a[1], xr, hi, 32, 1.0f);
    load_afrag_f32(a[2], sr, hi, 0, inv);   // agg = s * 1/max(cnt,1)
    load_afrag_f32(a[3], sr, hi, 32, inv);

#pragma unroll
    for (int nt = 0; nt < 4; ++nt) {
      v8f c = {};
#pragma unroll
      for (int kc = 0; kc < 4; ++kc) c = wmma_bf16(a[kc], bu1[kc * 4 + nt], c);
#pragma unroll
      for (int r = 0; r < 8; ++r)
        hbuf[wv][(r + 8 * hi) * 64 + nt * 16 + m] =
            (__bf16)gelu_fast(c[r] + bias1[nt]);
    }
    asm volatile("s_wait_dscnt 0" ::: "memory");

    v16bf a20, a21;
    load_afrag_lds(a20, &hbuf[wv][m * 64], hi, 0);
    load_afrag_lds(a21, &hbuf[wv][m * 64], hi, 32);

#pragma unroll
    for (int nt = 0; nt < 4; ++nt) {
      v8f c = {};
      c = wmma_bf16(a21, bu2[4 + nt], c);
      c = wmma_bf16(a20, bu2[nt], c);
#pragma unroll
      for (int r = 0; r < 8; ++r) {
        const int rowg = tile * 16 + r + 8 * hi;
        if (rowg < N)
          out[(long long)rowg * 64 + nt * 16 + m] = gelu_fast(c[r] + bias2[nt]);
      }
    }
  }
}

// ---------------------------------------------------------------------------
extern "C" void kernel_launch(void* const* d_in, const int* in_sizes, int n_in,
                              void* d_out, int out_size, void* d_ws, size_t ws_size,
                              hipStream_t stream) {
  const float* x     = (const float*)d_in[0];
  const int*   edges = (const int*)d_in[1];
  const float* ew    = (const float*)d_in[2];
  const float* pbn1  = (const float*)d_in[3];
  const float* pw1   = (const float*)d_in[4];
  const float* pc1   = (const float*)d_in[5];
  const float* pbn2  = (const float*)d_in[6];
  const float* pw2   = (const float*)d_in[7];
  const float* pc2   = (const float*)d_in[8];
  const float* ubn1  = (const float*)d_in[9];
  const float* uw1   = (const float*)d_in[10];
  const float* uc1   = (const float*)d_in[11];
  const float* ubn2  = (const float*)d_in[12];
  const float* uw2   = (const float*)d_in[13];
  const float* uc2   = (const float*)d_in[14];

  const int N = in_sizes[0] / 64;   // 100000
  const int E = in_sizes[2];        // 1600000

  // workspace carve-up
  char* w = (char*)d_ws;
  __bf16* W1f = (__bf16*)(w);                  // 64x64 bf16
  __bf16* W2f = W1f + 64 * 64;
  __bf16* U1f = W2f + 64 * 64;                 // 128x64 bf16
  __bf16* U2f = U1f + 128 * 64;
  float* c1f  = (float*)(U2f + 64 * 64);
  float* c2f  = c1f + 64;
  float* u1cf = c2f + 64;
  float* u2cf = u1cf + 64;
  float* s    = (float*)(w + 65536);           // [N,64] f32
  float* cnt  = s + (long long)N * 64;         // [N] f32
  float* out  = (float*)d_out;

  prep_kernel<<<1, 256, 0, stream>>>(pbn1, pw1, pc1, pbn2, pw2, pc2,
                                     ubn1, uw1, uc1, ubn2, uw2, uc2,
                                     W1f, c1f, W2f, c2f, U1f, u1cf, U2f, u2cf);
  init_kernel<<<1024, 256, 0, stream>>>(s, cnt, (long long)N * 64, (long long)N);
  edge_kernel<<<2048, 256, 0, stream>>>(x, edges, ew, W1f, c1f, W2f, c2f,
                                        s, cnt, E);
  node_kernel<<<1024, 256, 0, stream>>>(x, s, cnt, U1f, u1cf, U2f, u2cf,
                                        out, N);
}
